// GraphAttentionLayerV2_79328045957508
// MI455X (gfx1250) — matrix operand order repaired and checked by reading, not compiled, not run on hardware
//
#include <hip/hip_runtime.h>

// ---------------------------------------------------------------------------
// GAT layer, MI455X (gfx1250, wave32, WMMA + async global->LDS staging).
// Math: softmax shift-invariance kills the Wh_l branch entirely; attention
// becomes  h'[i,:] = (mask_i @ (P*Wh_r)) / (mask_i @ P)  -> one masked f16
// WMMA GEMM per (b,head) with the denominator folded in as an extra N-tile.
// Hot-loop staging uses GLOBAL_LOAD_ASYNC_TO_LDS_B128 (ASYNCcnt) from
// pre-laid-out f16 buffers (mask, V_aug transposed), so the j-loop is
// async-copy -> s_wait_asynccnt -> barrier -> ds_load_b128 -> v_wmma.
// ---------------------------------------------------------------------------

typedef __attribute__((ext_vector_type(16))) _Float16 v16h;
typedef __attribute__((ext_vector_type(8)))  _Float16 v8h;
typedef __attribute__((ext_vector_type(8)))  float    v8f;
typedef __attribute__((ext_vector_type(4)))  int      v4i;

#define GAT_B     4
#define GAT_N     2048
#define GAT_F     512
#define GAT_BH    32          // B * H
#define GAT_SLOPE 0.2f

// Build a 16-half WMMA fragment from two 16-byte-aligned LDS segments.
__device__ __forceinline__ v16h make16(const _Float16* p0, const _Float16* p1) {
  v8h a = *(const v8h*)p0;
  v8h b = *(const v8h*)p1;
  v16h r;
#pragma unroll
  for (int i = 0; i < 8; ++i) { r[i] = a[i]; r[i + 8] = b[i]; }
  return r;
}

// 32-bit LDS offset of a generic __shared__ pointer (low 32 bits of flat addr).
__device__ __forceinline__ unsigned lds_off(const void* p) {
  return (unsigned)(unsigned long long)p;
}

// Per-lane async copy of 16 bytes global -> LDS (tracked by ASYNCcnt).
__device__ __forceinline__ void async_b128(unsigned ldsaddr, unsigned voff,
                                           const void* sbase) {
  asm volatile("global_load_async_to_lds_b128 %0, %1, %2"
               :
               : "v"(ldsaddr), "v"(voff), "s"((unsigned long long)sbase)
               : "memory");
}

__device__ __forceinline__ void wait_async0() {
  asm volatile("s_wait_asynccnt 0x0" ::: "memory");
}

// ---------------------------------------------------------------------------
// K0: mask[b][i][j] = f16(adj > 0).  Vectorized: 8 elements / thread.
// ---------------------------------------------------------------------------
__global__ __launch_bounds__(256) void k_mask(const int* __restrict__ adj,
                                              _Float16* __restrict__ mask) {
  size_t g = (size_t)blockIdx.x * 256 + threadIdx.x;   // group of 8
  const v4i* src = (const v4i*)(adj + g * 8);
  v4i a0 = src[0], a1 = src[1];
  v8h m;
#pragma unroll
  for (int i = 0; i < 4; ++i) m[i]     = a0[i] > 0 ? (_Float16)1.0f : (_Float16)0.0f;
#pragma unroll
  for (int i = 0; i < 4; ++i) m[4 + i] = a1[i] > 0 ? (_Float16)1.0f : (_Float16)0.0f;
  *(v8h*)(mask + g * 8) = m;
}

// ---------------------------------------------------------------------------
// K1: Whr_f16[8192][512] = f16( leaky_relu( h[8192][512] @ Wr[512][512] ) )
// Block tile 128(M) x 64(N), K-step 32, 8 waves -> each wave one 16-row strip.
// ---------------------------------------------------------------------------
__global__ __launch_bounds__(256) void k_gemm_whr(const float* __restrict__ h,
                                                  const float* __restrict__ W,
                                                  _Float16* __restrict__ whr) {
  __shared__ _Float16 lA[128 * 40];   // [row][k], stride 40 halves (pad)
  __shared__ _Float16 lBT[64 * 40];   // transposed [n][k]
  const int tid = threadIdx.x;
  const int wv  = tid >> 5;
  const int ln  = tid & 31;
  const int m0  = blockIdx.x * 128;
  const int n0  = blockIdx.y * 64;
  const int khi = ln >> 4;
  const int nc  = ln & 15;

  v8f acc[4] = {};

  for (int k0 = 0; k0 < GAT_F; k0 += 32) {
    for (int idx = tid; idx < 128 * 32; idx += 256) {
      int r = idx >> 5, c = idx & 31;
      lA[r * 40 + c] = (_Float16)h[(m0 + r) * GAT_F + k0 + c];
    }
    for (int idx = tid; idx < 32 * 64; idx += 256) {
      int kk = idx >> 6, nn = idx & 63;
      lBT[nn * 40 + kk] = (_Float16)W[(k0 + kk) * GAT_F + n0 + nn];
    }
    __syncthreads();

    const int mrow = wv * 16 + nc;
    v16h af = make16(&lA[mrow * 40 + khi * 8], &lA[mrow * 40 + 16 + khi * 8]);
#pragma unroll
    for (int nt = 0; nt < 4; ++nt) {
      const _Float16* bp = &lBT[(nt * 16 + nc) * 40 + khi * 8];
      v16h bf = make16(bp, bp + 16);
      acc[nt] = __builtin_amdgcn_wmma_f32_16x16x32_f16(
          false, af, false, bf, (short)0, acc[nt], false, false);
    }
    __syncthreads();
  }

  // Epilogue: leaky-relu, store f16. C layout: VGPR r -> row 8*khi+r, lane&15 -> col.
#pragma unroll
  for (int nt = 0; nt < 4; ++nt) {
#pragma unroll
    for (int r = 0; r < 8; ++r) {
      int row = m0 + wv * 16 + khi * 8 + r;
      int col = n0 + nt * 16 + nc;
      float v = acc[nt][r];
      v = v > 0.f ? v : GAT_SLOPE * v;
      whr[row * GAT_F + col] = (_Float16)v;
    }
  }
}

// ---------------------------------------------------------------------------
// K2: er[c] = sum_k whr[c*64+k] * a[k]   (c = bh*2048 + n2, 65536 chunks)
// ---------------------------------------------------------------------------
__global__ __launch_bounds__(256) void k_er(const _Float16* __restrict__ whr,
                                            const float* __restrict__ a,
                                            float* __restrict__ er) {
  const int c  = blockIdx.x * 8 + (threadIdx.x >> 5);
  const int ln = threadIdx.x & 31;
  const _Float16* row = whr + (size_t)c * 64;
  float s = (float)row[ln] * a[ln] + (float)row[ln + 32] * a[ln + 32];
#pragma unroll
  for (int off = 16; off > 0; off >>= 1) s += __shfl_xor(s, off, 32);
  if (ln == 0) er[c] = s;
}

// ---------------------------------------------------------------------------
// K3: rowmax[bh] = max_j er[bh][j]
// ---------------------------------------------------------------------------
__global__ __launch_bounds__(256) void k_rowmax(const float* __restrict__ er,
                                                float* __restrict__ rmax) {
  __shared__ float sm[256];
  const float* row = er + blockIdx.x * GAT_N;
  float m = -3.4e38f;
  for (int j = threadIdx.x; j < GAT_N; j += 256) m = fmaxf(m, row[j]);
  sm[threadIdx.x] = m;
  __syncthreads();
  for (int s = 128; s > 0; s >>= 1) {
    if (threadIdx.x < s) sm[threadIdx.x] = fmaxf(sm[threadIdx.x], sm[threadIdx.x + s]);
    __syncthreads();
  }
  if (threadIdx.x == 0) rmax[blockIdx.x] = sm[0];
}

// ---------------------------------------------------------------------------
// K4: vaugT[bh][k][j]  (k < 64: P*whr, k==64: P, 65..79: 0),
//     P = exp(er[c] - rowmax[bh]), c = bh*2048 + j.
// Transposed layout so K5 can async-copy contiguous column segments.
// ---------------------------------------------------------------------------
__global__ __launch_bounds__(256) void k_scalev(const _Float16* __restrict__ whr,
                                                const float* __restrict__ er,
                                                const float* __restrict__ rmax,
                                                _Float16* __restrict__ vaugT) {
  const int base = blockIdx.x * 16;                    // chunk base
  for (int idx = threadIdx.x; idx < 16 * 80; idx += 256) {
    int cl = idx / 80, k = idx % 80;
    int c  = base + cl;
    int bh = c >> 11;
    int j  = c & (GAT_N - 1);
    float P = expf(er[c] - rmax[bh]);
    float v;
    if (k < 64)       v = P * (float)whr[(size_t)c * 64 + k];
    else if (k == 64) v = P;
    else              v = 0.f;
    vaugT[((size_t)bh * 80 + k) * GAT_N + j] = (_Float16)v;
  }
}

// ---------------------------------------------------------------------------
// K5: per (b,h):  C[2048][80] = mask_b[2048x2048] @ V_aug ;
//     out[i,k] = elu( C[i,k] / max(C[i,64], tiny) ).
// Block: 128 i-rows x 80 cols, 8 waves (16 rows each), K-step 32 over j.
// All staging via global_load_async_to_lds_b128 from f16 buffers laid out
// exactly as the LDS tiles need them (mask row-major, V transposed).
// ---------------------------------------------------------------------------
__global__ __launch_bounds__(256) void k_attn(const _Float16* __restrict__ mask,
                                              const _Float16* __restrict__ vaugT,
                                              float* __restrict__ out) {
  __shared__ _Float16 lA[128 * 40];   // mask tile [i][j], stride 40 halves
  __shared__ _Float16 lBT[80 * 40];   // V tile [col][j], stride 40 halves
  const int bh  = blockIdx.y;
  const int b   = bh >> 3;
  const int i0  = blockIdx.x * 128;
  const int tid = threadIdx.x;
  const int wv  = tid >> 5;
  const int ln  = tid & 31;
  const int khi = ln >> 4;
  const int nc  = ln & 15;

  const _Float16* maskb = mask  + (size_t)b  * GAT_N * GAT_N;
  const _Float16* vbT   = vaugT + (size_t)bh * 80 * GAT_N;
  const unsigned lAbase  = lds_off(&lA[0]);
  const unsigned lBTbase = lds_off(&lBT[0]);

  v8f acc[5] = {};

  for (int j0 = 0; j0 < GAT_N; j0 += 32) {
    // ---- async stage mask tile: 128 rows x 4 x 16B segments ----
    for (int idx = tid; idx < 128 * 4; idx += 256) {
      int r = idx >> 2, seg = idx & 3;
      unsigned la = lAbase + (unsigned)(r * 40 + seg * 8) * 2u;
      unsigned go = (unsigned)(((i0 + r) * GAT_N + j0 + seg * 8) * 2);
      async_b128(la, go, maskb);
    }
    // ---- async stage V tile: 80 cols x 4 x 16B segments ----
    for (int idx = tid; idx < 80 * 4; idx += 256) {
      int cc = idx >> 2, seg = idx & 3;
      unsigned la = lBTbase + (unsigned)(cc * 40 + seg * 8) * 2u;
      unsigned go = (unsigned)((cc * GAT_N + j0 + seg * 8) * 2);
      async_b128(la, go, vbT);
    }
    wait_async0();
    __syncthreads();

    const int mrow = wv * 16 + nc;
    v16h af = make16(&lA[mrow * 40 + khi * 8], &lA[mrow * 40 + 16 + khi * 8]);
#pragma unroll
    for (int nt = 0; nt < 5; ++nt) {
      const _Float16* bp = &lBT[(nt * 16 + nc) * 40 + khi * 8];
      v16h bf = make16(bp, bp + 16);
      acc[nt] = __builtin_amdgcn_wmma_f32_16x16x32_f16(
          false, af, false, bf, (short)0, acc[nt], false, false);
    }
    __syncthreads();
  }

  // Epilogue: divide by denominator (col 64, held by lanes with nc==0), elu, store.
#pragma unroll
  for (int r = 0; r < 8; ++r) {
    float S   = __shfl(acc[4][r], khi << 4, 32);
    float inv = 1.0f / fmaxf(S, 1e-30f);
    int   row = i0 + wv * 16 + khi * 8 + r;
    size_t ob = ((size_t)bh * GAT_N + row) * 64;
#pragma unroll
    for (int nt = 0; nt < 4; ++nt) {
      float x = acc[nt][r] * inv;
      x = x > 0.f ? x : (expf(x) - 1.0f);
      out[ob + nt * 16 + nc] = x;
    }
  }
}

// ---------------------------------------------------------------------------
extern "C" void kernel_launch(void* const* d_in, const int* in_sizes, int n_in,
                              void* d_out, int out_size, void* d_ws, size_t ws_size,
                              hipStream_t stream) {
  (void)in_sizes; (void)n_in; (void)out_size; (void)ws_size;
  const float* h   = (const float*)d_in[0];
  const int*   adj = (const int*)d_in[1];
  // d_in[2] (W_l) is mathematically dead: softmax shift-invariance cancels el.
  const float* Wr  = (const float*)d_in[3];
  const float* a   = (const float*)d_in[4];
  float* out = (float*)d_out;

  char* ws = (char*)d_ws;
  _Float16* whr   = (_Float16*)(ws);                           //  8,388,608 B
  float*    er    = (float*)   (ws + 8388608);                 //    262,144 B
  float*    rmax  = (float*)   (ws + 8388608 + 262144);        //        128 B
  _Float16* vaugT = (_Float16*)(ws + 8388608 + 262144 + 128);  // 10,485,760 B
  _Float16* msk   = (_Float16*)(ws + 8388608 + 262144 + 128 + 10485760); // 33,554,432 B

  k_mask    <<<dim3(8192),  256, 0, stream>>>(adj, msk);
  k_gemm_whr<<<dim3(64, 8), 256, 0, stream>>>(h, Wr, whr);
  k_er      <<<dim3(8192),  256, 0, stream>>>(whr, a, er);
  k_rowmax  <<<dim3(GAT_BH),256, 0, stream>>>(er, rmax);
  k_scalev  <<<dim3(4096),  256, 0, stream>>>(whr, er, rmax, vaugT);
  k_attn    <<<dim3(16, GAT_BH), 256, 0, stream>>>(msk, vaugT, out);
}
